// Transformer_61375082660266
// MI455X (gfx1250) — matrix operand order repaired and checked by reading, not compiled, not run on hardware
//
#include <hip/hip_runtime.h>
#include <hip/hip_bf16.h>

// Problem constants (from reference): L=8, V=32000, D=1024, DH=64, T=512, H=16, B=8
#define L_    8
#define V_    32000
#define D_    1024
#define DH_   64
#define T_    512
#define H_    16
#define B_    8
#define DFF_  4096
#define MROWS (B_ * T_)   // 4096

typedef __bf16 bf16;
typedef __attribute__((ext_vector_type(16))) __bf16 v16bf;
typedef __attribute__((ext_vector_type(8)))  float  v8f;

union ABFrag { v16bf v; uint4 q[2]; };

__device__ __forceinline__ v8f wmma_bf16(v16bf a, v16bf b, v8f c) {
  // 8 args: (neg_a, A, neg_b, B, c_mod, C, reuse_a, reuse_b)
  return __builtin_amdgcn_wmma_f32_16x16x32_bf16(false, a, false, b, (short)0, c, false, false);
}

// ---------------------------------------------------------------------------
// CDNA5 async global->LDS copy (ASYNCcnt-tracked), guarded so compile can't fail
// Builtin signature (from clang diagnostic): (int4 as(1)*, int4 as(3)*, Ii, Ii)
// where int4 is the GCC-style vector_size(16) int and as(1) prints as __device__.
// ---------------------------------------------------------------------------
#if defined(__AMDGCN__) && __has_builtin(__builtin_amdgcn_global_load_async_to_lds_b128)
#define HAS_ASYNC_LDS 1
#else
#define HAS_ASYNC_LDS 0
#endif

#if HAS_ASYNC_LDS
typedef int v4i_vec __attribute__((vector_size(16)));
typedef __attribute__((address_space(1))) v4i_vec g_v4i;
typedef __attribute__((address_space(3))) v4i_vec l_v4i;
__device__ __forceinline__ void async_cp16(const bf16* g, bf16* l) {
  __builtin_amdgcn_global_load_async_to_lds_b128((g_v4i*)g, (l_v4i*)l, 0, 0);
}
#endif

__device__ __forceinline__ void wait_async0() {
#if defined(__AMDGCN__)
#if __has_builtin(__builtin_amdgcn_s_wait_asynccnt)
  __builtin_amdgcn_s_wait_asynccnt(0);
#elif HAS_ASYNC_LDS
  asm volatile("s_wait_asynccnt 0" ::: "memory");
#endif
#endif
}

// ---------------------------------------------------------------------------
// Repack kernels: fp32 weights -> bf16 in [N][K] (B-transposed) layout
// ---------------------------------------------------------------------------

// Wq/Wk/Wv[l]: [H][D][DH] fp32 -> out[n][d] bf16 with n = h*64 + e
__global__ __launch_bounds__(256) void repack_qkv_kernel(
    const float* __restrict__ in, bf16* __restrict__ out) {
  int i = blockIdx.x * 256 + threadIdx.x;      // over D_*D_ outputs
  int n = i >> 10, d = i & (D_ - 1);
  int hh = n >> 6, e = n & 63;
  out[i] = (bf16)in[((size_t)hh * D_ + d) * DH_ + e];
}

// in: [K][N] fp32 row-major -> out: [N][K] bf16
__global__ __launch_bounds__(256) void repack_T_kernel(
    const float* __restrict__ in, bf16* __restrict__ out, int K, int N) {
  size_t i = (size_t)blockIdx.x * 256 + threadIdx.x;  // over (size_t)N*K outputs
  size_t n = i / (size_t)K, kk = i % (size_t)K;
  out[i] = (bf16)in[kk * (size_t)N + n];
}

// ---------------------------------------------------------------------------
// Embedding + sinusoidal positions: h[m][d] = emb[x[m]][d] + sin(t * 10000^(-2d/D))
// ---------------------------------------------------------------------------
__global__ __launch_bounds__(256) void embed_kernel(
    const int* __restrict__ x, const float* __restrict__ emb,
    float* __restrict__ h, bf16* __restrict__ hb) {
  const int m = blockIdx.x, tid = threadIdx.x;
  const int t = m & (T_ - 1);
  const int tok = x[m];
  for (int i = 0; i < 4; ++i) {
    int d = tid + i * 256;
    float inv = __expf(-9.210340371976184f * (2.0f * (float)d) / (float)D_); // 10000^(-2d/D)
    float v = emb[(size_t)tok * D_ + d] + sinf((float)t * inv);
    h[(size_t)m * D_ + d]  = v;
    hb[(size_t)m * D_ + d] = (bf16)v;
  }
}

// ---------------------------------------------------------------------------
// bf16 WMMA GEMM: C[M][N] = A[M][K] (row-major bf16) * BT[N][K] (bf16)
// Block: 256 threads = 8 waves; block tile 128x128; BK=64 (2 WMMA k-steps/stage).
// Double-buffered LDS; tiles staged with GLOBAL_LOAD_ASYNC_TO_LDS (ASYNCcnt)
// when available so the next tile streams in during the current tile's WMMAs.
// Wave (wm in 0..3, wn in 0..1) computes 32x64 = 2x4 WMMA tiles.
// OUTMODE: 0 = f32 row-major, 1 = bf16 row-major,
//          2 = bf16 [B,H,T,DH] (q/k layout), 3 = bf16 [B,H,DH,T] (v^T layout)
// M,N multiples of 128; K multiple of 64 (true for all uses here).
// ---------------------------------------------------------------------------
#define BK_     64
#define LPITCH_ 72   // row pitch in bf16 elems (144B: 16B-aligned, bank-spread)

template<int OUTMODE, bool BIAS, bool RES, bool LEAKY>
__global__ __launch_bounds__(256) void gemm_bf16_kernel(
    const bf16* __restrict__ A, const bf16* __restrict__ BT,
    void* __restrict__ Cout, const float* __restrict__ bias,
    const float* __restrict__ resid, int M, int N, int K) {
  __shared__ __align__(16) bf16 As[2][128][LPITCH_];
  __shared__ __align__(16) bf16 Bs[2][128][LPITCH_];

  const int tid  = threadIdx.x;
  const int lane = tid & 31, wave = tid >> 5;
  const int l16  = lane & 15, lh = lane >> 4;
  const int wm   = wave >> 1, wn = wave & 1;
  const int mblock = blockIdx.y * 128, nblock = blockIdx.x * 128;

  v8f acc[2][4];
  for (int i = 0; i < 2; ++i)
    for (int j = 0; j < 4; ++j) acc[i][j] = (v8f)0.0f;

  // Staging map: thread copies 64B (4x16B) per matrix: row = tid/2, 32-elem half.
  const int ldrow = tid >> 1;
  const int ldcol = (tid & 1) * 32;
  const bf16* agBase = A  + (size_t)(mblock + ldrow) * K + ldcol;
  const bf16* bgBase = BT + (size_t)(nblock + ldrow) * K + ldcol;

  auto issue = [&](int kt, int buf) {
    const bf16* ag = agBase + kt * BK_;
    const bf16* bg = bgBase + kt * BK_;
    bf16* al = &As[buf][ldrow][ldcol];
    bf16* bl = &Bs[buf][ldrow][ldcol];
#if HAS_ASYNC_LDS
    for (int j = 0; j < 4; ++j) {
      async_cp16(ag + 8 * j, al + 8 * j);
      async_cp16(bg + 8 * j, bl + 8 * j);
    }
#else
    for (int j = 0; j < 4; ++j) {
      *(uint4*)(al + 8 * j) = *(const uint4*)(ag + 8 * j);
      *(uint4*)(bl + 8 * j) = *(const uint4*)(bg + 8 * j);
    }
#endif
  };

  const int KT = K / BK_;
  issue(0, 0);
  for (int kt = 0; kt < KT; ++kt) {
    const int buf = kt & 1;
    wait_async0();        // our stage-kt copies have landed in LDS
    __syncthreads();      // everyone's landed; everyone done reading buf^1
    if (kt + 1 < KT) issue(kt + 1, buf ^ 1);   // streams during WMMAs below

    for (int ks = 0; ks < 2; ++ks) {           // two 16x16x32 k-steps per stage
      ABFrag a[2], b[4];
      for (int mt = 0; mt < 2; ++mt) {
        // A frag: lane holds row (l16), contiguous 16 K-elems at lh*16
        const bf16* ap = &As[buf][wm * 32 + mt * 16 + l16][ks * 32 + lh * 16];
        a[mt].q[0] = *(const uint4*)ap;
        a[mt].q[1] = *(const uint4*)(ap + 8);
      }
      for (int nt = 0; nt < 4; ++nt) {
        // B frag: lane holds col (l16), contiguous 16 K-elems at lh*16
        const bf16* bp = &Bs[buf][wn * 64 + nt * 16 + l16][ks * 32 + lh * 16];
        b[nt].q[0] = *(const uint4*)bp;
        b[nt].q[1] = *(const uint4*)(bp + 8);
      }
      for (int mt = 0; mt < 2; ++mt)
        for (int nt = 0; nt < 4; ++nt)
          acc[mt][nt] = wmma_bf16(a[mt].v, b[nt].v, acc[mt][nt]);
    }
  }

  // Epilogue. C layout: VGPR r -> row base+lh*8+r, col = l16 within 16-wide tile.
  for (int mt = 0; mt < 2; ++mt) {
    for (int nt = 0; nt < 4; ++nt) {
      const int n = nblock + wn * 64 + nt * 16 + l16;
      const float bval = BIAS ? bias[n] : 0.0f;
      for (int r = 0; r < 8; ++r) {
        const int m = mblock + wm * 32 + mt * 16 + lh * 8 + r;
        float c = acc[mt][nt][r] + bval;
        if (LEAKY) c = c > 0.0f ? c : 0.01f * c;
        if (RES)   c += resid[(size_t)m * N + n];
        if (OUTMODE == 0) {
          ((float*)Cout)[(size_t)m * N + n] = c;
        } else if (OUTMODE == 1) {
          ((bf16*)Cout)[(size_t)m * N + n] = (bf16)c;
        } else if (OUTMODE == 2) {
          const int bb = m >> 9, t = m & (T_ - 1), hh = n >> 6, e = n & 63;
          ((bf16*)Cout)[((size_t)(bb * H_ + hh) * T_ + t) * DH_ + e] = (bf16)c;
        } else { // OUTMODE == 3 : V^T [B,H,DH,T]
          const int bb = m >> 9, t = m & (T_ - 1), hh = n >> 6, e = n & 63;
          ((bf16*)Cout)[((size_t)(bb * H_ + hh) * DH_ + e) * T_ + t] = (bf16)c;
        }
      }
    }
  }
}

// ---------------------------------------------------------------------------
// Fused causal flash attention (per (b,h), 64-query tile; 4 waves x 16 rows).
// q,k: bf16 [B,H,T,DH]; vt: bf16 [B,H,DH,T]; out: bf16 [B,T,H*DH].
// Layouts chosen so K/V B-fragments are direct contiguous global b128 loads.
// Reference semantics: no pre-softmax scale; POST-softmax * sqrt(DH)=8.
// ---------------------------------------------------------------------------
__global__ __launch_bounds__(128) void attention_kernel(
    const bf16* __restrict__ q, const bf16* __restrict__ k,
    const bf16* __restrict__ vt, bf16* __restrict__ attn_out) {
  __shared__ __align__(16) bf16 Ps[4][16][64];   // per-wave P staging (C->A relayout)

  const int lane = threadIdx.x & 31, wave = threadIdx.x >> 5;
  const int l16 = lane & 15, lh = lane >> 4;
  const int qblk = blockIdx.x, hh = blockIdx.y, bb = blockIdx.z;
  const int qbase = qblk * 64 + wave * 16;

  const bf16* qp = q  + (size_t)(bb * H_ + hh) * T_ * DH_;
  const bf16* kp = k  + (size_t)(bb * H_ + hh) * T_ * DH_;
  const bf16* vp = vt + (size_t)(bb * H_ + hh) * DH_ * T_;

  // Q A-fragments (K dim = DH = 64 -> 2 k-steps of 32), loaded once.
  v16bf qf[2];
  for (int ks = 0; ks < 2; ++ks) {
    ABFrag t;
    const bf16* p = qp + (size_t)(qbase + l16) * DH_ + ks * 32 + lh * 16;
    t.q[0] = *(const uint4*)p;
    t.q[1] = *(const uint4*)(p + 8);
    qf[ks] = t.v;
  }

  v8f o[4];
  for (int nt = 0; nt < 4; ++nt) o[nt] = (v8f)0.0f;
  float mrow[8], lrow[8];
  for (int r = 0; r < 8; ++r) { mrow[r] = -3.0e38f; lrow[r] = 0.0f; }

  for (int kb = 0; kb <= qblk; ++kb) {            // causal: skip blocks past diagonal
    // S = Q * K^T over a 64-key block (4 n-tiles x 2 k-steps)
    v8f s[4];
    for (int nt = 0; nt < 4; ++nt) s[nt] = (v8f)0.0f;
    for (int ks = 0; ks < 2; ++ks) {
      for (int nt = 0; nt < 4; ++nt) {
        ABFrag kf;  // B frag straight from global: row = key s, contiguous e chunk
        const bf16* p = kp + (size_t)(kb * 64 + nt * 16 + l16) * DH_ + ks * 32 + lh * 16;
        kf.q[0] = *(const uint4*)p;
        kf.q[1] = *(const uint4*)(p + 8);
        s[nt] = wmma_bf16(qf[ks], kf.v, s[nt]);
      }
    }
    if (kb == qblk) {                              // diagonal block: elementwise mask
      for (int nt = 0; nt < 4; ++nt)
        for (int r = 0; r < 8; ++r) {
          int scol = kb * 64 + nt * 16 + l16;
          int trow = qbase + lh * 8 + r;
          if (scol > trow) s[nt][r] = -3.0e38f;
        }
    }

    // Online softmax: row reductions across the 16-lane half-wave
    float mloc[8];
    for (int r = 0; r < 8; ++r)
      mloc[r] = fmaxf(fmaxf(s[0][r], s[1][r]), fmaxf(s[2][r], s[3][r]));
    for (int off = 8; off >= 1; off >>= 1)
      for (int r = 0; r < 8; ++r)
        mloc[r] = fmaxf(mloc[r], __shfl_xor(mloc[r], off, 32));

    float psc[8];
    for (int r = 0; r < 8; ++r) {
      float mnew = fmaxf(mrow[r], mloc[r]);
      psc[r] = __expf(mrow[r] - mnew);
      mrow[r] = mnew;
      lrow[r] *= psc[r];
    }
    for (int nt = 0; nt < 4; ++nt)
      for (int r = 0; r < 8; ++r) o[nt][r] *= psc[r];

    float rs[8];
    for (int r = 0; r < 8; ++r) rs[r] = 0.0f;
    for (int nt = 0; nt < 4; ++nt)
      for (int r = 0; r < 8; ++r) {
        float p = __expf(s[nt][r] - mrow[r]);
        rs[r] += p;
        Ps[wave][lh * 8 + r][nt * 16 + l16] = (bf16)p;   // C-layout -> LDS row-major
      }
    for (int off = 8; off >= 1; off >>= 1)
      for (int r = 0; r < 8; ++r) rs[r] += __shfl_xor(rs[r], off, 32);
    for (int r = 0; r < 8; ++r) lrow[r] += rs[r];

    __syncthreads();   // uniform across block (all waves share qblk loop bound)

    // O += P * V  (A frags of P from LDS; V B-frags straight from v^T global)
    for (int ks = 0; ks < 2; ++ks) {
      ABFrag pf;
      const bf16* pp = &Ps[wave][l16][ks * 32 + lh * 16];
      pf.q[0] = *(const uint4*)pp;
      pf.q[1] = *(const uint4*)(pp + 8);
      for (int nt = 0; nt < 4; ++nt) {
        ABFrag vf;  // row = output dim e, contiguous key chunk
        const bf16* p = vp + (size_t)(nt * 16 + l16) * T_ + kb * 64 + ks * 32 + lh * 16;
        vf.q[0] = *(const uint4*)p;
        vf.q[1] = *(const uint4*)(p + 8);
        o[nt] = wmma_bf16(pf.v, vf.v, o[nt]);
      }
    }
    __syncthreads();
  }

  // Finalize: divide by row sum, apply the reference's POST-softmax sqrt(DH)=8
  for (int nt = 0; nt < 4; ++nt)
    for (int r = 0; r < 8; ++r) {
      int trow = qbase + lh * 8 + r;
      int e = nt * 16 + l16;
      float val = o[nt][r] / lrow[r] * 8.0f;
      attn_out[(size_t)(bb * T_ + trow) * D_ + hh * DH_ + e] = (bf16)val;
    }
}

// ---------------------------------------------------------------------------
// LayerNorm over D=1024 (1 block / row), writes f32 h and bf16 copy
// ---------------------------------------------------------------------------
__global__ __launch_bounds__(256) void layernorm_kernel(
    const float* __restrict__ in, const float* __restrict__ gamma,
    const float* __restrict__ beta, float* __restrict__ hout,
    bf16* __restrict__ hbout) {
  __shared__ float red[256];
  const int m = blockIdx.x, tid = threadIdx.x;
  const float* row = in + (size_t)m * D_;

  float x[4], s = 0.0f;
  for (int i = 0; i < 4; ++i) { x[i] = row[tid + i * 256]; s += x[i]; }
  red[tid] = s; __syncthreads();
  for (int o = 128; o >= 1; o >>= 1) { if (tid < o) red[tid] += red[tid + o]; __syncthreads(); }
  const float mean = red[0] * (1.0f / D_);
  __syncthreads();

  float ss = 0.0f;
  for (int i = 0; i < 4; ++i) { float d = x[i] - mean; ss += d * d; }
  red[tid] = ss; __syncthreads();
  for (int o = 128; o >= 1; o >>= 1) { if (tid < o) red[tid] += red[tid + o]; __syncthreads(); }
  const float rstd = rsqrtf(red[0] * (1.0f / D_) + 1e-5f);

  for (int i = 0; i < 4; ++i) {
    int d = tid + i * 256;
    float v = (x[i] - mean) * rstd * gamma[d] + beta[d];
    hout[(size_t)m * D_ + d]  = v;
    hbout[(size_t)m * D_ + d] = (bf16)v;
  }
}

// ---------------------------------------------------------------------------
// Per-row log-softmax loss; deterministic two-stage reduction
// ---------------------------------------------------------------------------
__global__ __launch_bounds__(256) void loss_row_kernel(
    const float* __restrict__ logits, const int* __restrict__ y,
    float* __restrict__ rowloss) {
  __shared__ float red[256];
  const int m = blockIdx.x, tid = threadIdx.x;
  const float* row = logits + (size_t)m * V_;

  float mx = -3.0e38f;
  for (int i = tid; i < V_; i += 256) mx = fmaxf(mx, row[i]);
  red[tid] = mx; __syncthreads();
  for (int o = 128; o >= 1; o >>= 1) { if (tid < o) red[tid] = fmaxf(red[tid], red[tid + o]); __syncthreads(); }
  mx = red[0]; __syncthreads();

  float s = 0.0f;
  for (int i = tid; i < V_; i += 256) s += __expf(row[i] - mx);
  red[tid] = s; __syncthreads();
  for (int o = 128; o >= 1; o >>= 1) { if (tid < o) red[tid] += red[tid + o]; __syncthreads(); }
  if (tid == 0) rowloss[m] = row[y[m]] - mx - __logf(red[0]);   // log-prob of target
}

__global__ __launch_bounds__(256) void loss_reduce_kernel(
    const float* __restrict__ rowloss, float* __restrict__ loss) {
  __shared__ float red[256];
  float s = 0.0f;
  for (int i = threadIdx.x; i < MROWS; i += 256) s += rowloss[i];
  red[threadIdx.x] = s; __syncthreads();
  for (int o = 128; o >= 1; o >>= 1) { if (threadIdx.x < o) red[threadIdx.x] += red[threadIdx.x + o]; __syncthreads(); }
  if (threadIdx.x == 0) *loss = -red[0] * (1.0f / (float)MROWS);
}

// ---------------------------------------------------------------------------
// Host orchestration
// ---------------------------------------------------------------------------
extern "C" void kernel_launch(void* const* d_in, const int* in_sizes, int n_in,
                              void* d_out, int out_size, void* d_ws, size_t ws_size,
                              hipStream_t stream) {
  const int*   x    = (const int*)d_in[0];
  const int*   y    = (const int*)d_in[1];
  const float* emb  = (const float*)d_in[2];
  const float* Wq   = (const float*)d_in[3];
  const float* bq   = (const float*)d_in[4];
  const float* Wk   = (const float*)d_in[5];
  const float* bk   = (const float*)d_in[6];
  const float* Wv   = (const float*)d_in[7];
  const float* bv   = (const float*)d_in[8];
  const float* Wo   = (const float*)d_in[9];
  const float* bo   = (const float*)d_in[10];
  const float* ln1g = (const float*)d_in[11];
  const float* ln1b = (const float*)d_in[12];
  const float* W1   = (const float*)d_in[13];
  const float* b1   = (const float*)d_in[14];
  const float* W2   = (const float*)d_in[15];
  const float* b2   = (const float*)d_in[16];
  const float* ln2g = (const float*)d_in[17];
  const float* ln2b = (const float*)d_in[18];
  const float* Wu   = (const float*)d_in[19];
  const float* bu   = (const float*)d_in[20];

  float* logits = (float*)d_out;                          // [4096, 32000]
  float* loss   = logits + (size_t)MROWS * V_;            // scalar after logits

  char* ws = (char*)d_ws;
  size_t off = 0;
  auto alloc = [&](size_t bytes) -> void* {
    void* p = ws + off;
    off = (off + bytes + 255) & ~(size_t)255;
    return p;
  };
  float* h     = (float*)alloc((size_t)MROWS * D_ * 4);
  bf16*  hb    = (bf16*) alloc((size_t)MROWS * D_ * 2);
  float* tmp   = (float*)alloc((size_t)MROWS * D_ * 4);
  bf16*  qb    = (bf16*) alloc((size_t)MROWS * D_ * 2);
  bf16*  kbuf  = (bf16*) alloc((size_t)MROWS * D_ * 2);
  bf16*  vtb   = (bf16*) alloc((size_t)MROWS * D_ * 2);
  bf16*  attn  = (bf16*) alloc((size_t)MROWS * D_ * 2);
  bf16*  f1    = (bf16*) alloc((size_t)MROWS * DFF_ * 2);
  bf16*  wqT   = (bf16*) alloc((size_t)D_ * D_ * 2);
  bf16*  wkT   = (bf16*) alloc((size_t)D_ * D_ * 2);
  bf16*  wvT   = (bf16*) alloc((size_t)D_ * D_ * 2);
  bf16*  woT   = (bf16*) alloc((size_t)D_ * D_ * 2);
  bf16*  w1T   = (bf16*) alloc((size_t)D_ * DFF_ * 2);
  bf16*  w2T   = (bf16*) alloc((size_t)D_ * DFF_ * 2);
  bf16*  wuT   = (bf16*) alloc((size_t)D_ * V_ * 2);
  float* rowl  = (float*)alloc((size_t)MROWS * 4);

  embed_kernel<<<MROWS, 256, 0, stream>>>(x, emb, h, hb);
  repack_T_kernel<<<(unsigned)(((size_t)D_ * V_) / 256), 256, 0, stream>>>(Wu, wuT, D_, V_);

  const dim3 g_d(D_ / 128, MROWS / 128);      // (8, 32)
  const dim3 g_ff(DFF_ / 128, MROWS / 128);   // (32, 32)
  const dim3 g_v(V_ / 128, MROWS / 128);      // (250, 32)

  for (int l = 0; l < L_; ++l) {
    repack_qkv_kernel<<<D_ * D_ / 256, 256, 0, stream>>>(Wq + (size_t)l * H_ * D_ * DH_, wqT);
    repack_qkv_kernel<<<D_ * D_ / 256, 256, 0, stream>>>(Wk + (size_t)l * H_ * D_ * DH_, wkT);
    repack_qkv_kernel<<<D_ * D_ / 256, 256, 0, stream>>>(Wv + (size_t)l * H_ * D_ * DH_, wvT);
    repack_T_kernel<<<D_ * D_ / 256, 256, 0, stream>>>(Wo + (size_t)l * D_ * D_, woT, D_, D_);
    repack_T_kernel<<<D_ * DFF_ / 256, 256, 0, stream>>>(W1 + (size_t)l * D_ * DFF_, w1T, D_, DFF_);
    repack_T_kernel<<<D_ * DFF_ / 256, 256, 0, stream>>>(W2 + (size_t)l * DFF_ * D_, w2T, DFF_, D_);

    // q,k in [B,H,T,DH]; v written transposed [B,H,DH,T]
    gemm_bf16_kernel<2, true, false, false><<<g_d, 256, 0, stream>>>(hb, wqT, qb,   bq + l * D_, nullptr, MROWS, D_, D_);
    gemm_bf16_kernel<2, true, false, false><<<g_d, 256, 0, stream>>>(hb, wkT, kbuf, bk + l * D_, nullptr, MROWS, D_, D_);
    gemm_bf16_kernel<3, true, false, false><<<g_d, 256, 0, stream>>>(hb, wvT, vtb,  bv + l * D_, nullptr, MROWS, D_, D_);

    attention_kernel<<<dim3(T_ / 64, H_, B_), 128, 0, stream>>>(qb, kbuf, vtb, attn);

    // o @ Wo + bo + h  -> tmp ; LN1 -> h, hb
    gemm_bf16_kernel<0, true, true, false><<<g_d, 256, 0, stream>>>(attn, woT, tmp, bo + l * D_, h, MROWS, D_, D_);
    layernorm_kernel<<<MROWS, 256, 0, stream>>>(tmp, ln1g + l * D_, ln1b + l * D_, h, hb);

    // FFN: leaky_relu(h @ W1 + b1) -> f1 (bf16) ; f1 @ W2 + b2 + h -> tmp ; LN2
    gemm_bf16_kernel<1, true, false, true><<<g_ff, 256, 0, stream>>>(hb, w1T, f1, b1 + (size_t)l * DFF_, nullptr, MROWS, DFF_, D_);
    gemm_bf16_kernel<0, true, true, false><<<g_d, 256, 0, stream>>>(f1, w2T, tmp, b2 + l * D_, h, MROWS, D_, DFF_);
    layernorm_kernel<<<MROWS, 256, 0, stream>>>(tmp, ln2g + l * D_, ln2b + l * D_, h, hb);
  }

  // logits = h @ Wu + bu  (f32 straight into d_out), then deterministic loss
  gemm_bf16_kernel<0, true, false, false><<<g_v, 256, 0, stream>>>(hb, wuT, logits, bu, nullptr, MROWS, V_, D_);
  loss_row_kernel<<<MROWS, 256, 0, stream>>>(logits, y, rowl);
  loss_reduce_kernel<<<1, 256, 0, stream>>>(rowl, loss);
}
